// AugmentedMemoryScaledDotProductAttention_36524401885989
// MI455X (gfx1250) — compile-verified, matrix operand order checked by reading, not run on hardware
//
#include <hip/hip_runtime.h>
#include <hip/hip_bf16.h>

// ---------------------------------------------------------------------------
// Augmented-memory multi-head attention for MI455X (gfx1250, wave32, WMMA).
// Fused scores+mask+softmax+att@V kernel keeps a 32-query strip of the full
// score row (32 x 1040 fp32) in LDS (320KB/WGP on CDNA5), so the 273MB
// attention matrix is written to HBM exactly once and never re-read.
// Q tile staged to LDS via gfx1250 async load-to-LDS (builtin signature:
// (v4i as1*, v4i as3*, imm offset, imm cpol) — deduced from clang diagnostic).
// All GEMMs: v_wmma_f32_16x16x32_bf16, fp32 accumulate.
// ---------------------------------------------------------------------------

#define DEV_INLINE __device__ __forceinline__

typedef __attribute__((ext_vector_type(16))) __bf16 v16bf;
typedef __attribute__((ext_vector_type(8)))  __bf16 v8bf;
typedef __attribute__((ext_vector_type(8)))  float  v8f;
typedef __attribute__((ext_vector_type(4)))  int    v4i;

typedef __attribute__((address_space(1))) v4i as1_v4i;
typedef __attribute__((address_space(3))) v4i as3_v4i;

static constexpr int B_    = 8;
static constexpr int NQ    = 1024;
static constexpr int NK    = 1024;
static constexpr int H_    = 8;
static constexpr int DK    = 128;
static constexpr int DM    = 1024;       // D_MODEL == H*DK
static constexpr int MSLOT = 16;
static constexpr int NKM   = NK + MSLOT; // 1040
static constexpr int VTLD  = 1056;       // V^T row stride (zero padded)

// Fused-kernel LDS layout (dynamic LDS; 144000 B total, 2 WGs per 320KB WGP)
static constexpr int QTILE   = 32;
static constexpr int SSTRIDE = 1057;     // odd stride -> 16 frag rows hit distinct banks
static constexpr int QSTRIDE = 136;      // 272B rows -> conflict-free b128 frag reads
static constexpr int SMEM_S_BYTES = QTILE * SSTRIDE * 4;                 // 135296
static constexpr int SMEM_BYTES   = SMEM_S_BYTES + QTILE * QSTRIDE * 2;  // 144000

#if defined(__has_builtin)
#if __has_builtin(__builtin_amdgcn_global_load_async_to_lds_b128) && \
    __has_builtin(__builtin_amdgcn_s_wait_asynccnt)
#define HAVE_ASYNC_LDS 1
#endif
#endif

DEV_INLINE v8f wmma_bf16(v16bf a, v16bf b, v8f c) {
  return __builtin_amdgcn_wmma_f32_16x16x32_bf16(
      /*neg_a=*/false, a, /*neg_b=*/false, b,
      /*c_mod=*/(short)0, c, /*reuse_a=*/false, /*reuse_b=*/false);
}

// A/B fragment load from row-major bf16 (works for global or LDS pointers).
// Lane L<16: row=L, K elems {0..7},{16..23}; L>=16: row=L-16, {8..15},{24..31}.
DEV_INLINE v16bf frag_ld(const __bf16* __restrict__ mat, int ld,
                         int row_base, int row_max, int col) {
  const int lane = threadIdx.x & 31;
  int row = row_base + (lane & 15);
  row = row > row_max ? row_max : row;
  const int ko = (lane & 16) ? 8 : 0;
  const __bf16* p = mat + (size_t)row * ld + col + ko;
  v8bf lo = *reinterpret_cast<const v8bf*>(p);
  v8bf hi = *reinterpret_cast<const v8bf*>(p + 16);
  v16bf f;
#pragma unroll
  for (int i = 0; i < 8; ++i) { f[i] = lo[i]; f[i + 8] = hi[i]; }
  return f;
}

// A fragment from fp32 source (global or LDS), converted to bf16 on load;
// optional zero of the upper K-half for the K tail (1040 = 32*32 + 16).
DEV_INLINE v16bf frag_ld_f32(const float* __restrict__ mat, int ld,
                             int row_base, int col, bool hi_zero) {
  const int lane = threadIdx.x & 31;
  const int row = row_base + (lane & 15);
  const int ko = (lane & 16) ? 8 : 0;
  const float* p = mat + (size_t)row * ld + col + ko;
  v16bf f;
#pragma unroll
  for (int i = 0; i < 8; ++i) f[i] = (__bf16)p[i];
  if (hi_zero) {
#pragma unroll
    for (int i = 0; i < 8; ++i) f[i + 8] = (__bf16)0.0f;
  } else {
#pragma unroll
    for (int i = 0; i < 8; ++i) f[i + 8] = (__bf16)p[16 + i];
  }
  return f;
}

// ---------------------------------------------------------------------------
// Elementwise prep kernels
// ---------------------------------------------------------------------------
__global__ __launch_bounds__(256) void cvt_f32_bf16(const float* __restrict__ in,
                                                    __bf16* __restrict__ out,
                                                    unsigned int n) {
  unsigned int i = blockIdx.x * 256u + threadIdx.x;
  if (i < n) out[i] = (__bf16)in[i];
}

__global__ __launch_bounds__(256) void wt_cvt(const float* __restrict__ W,
                                              __bf16* __restrict__ WT) {
  unsigned int idx = blockIdx.x * 256u + threadIdx.x;  // 1M elements
  unsigned int o = idx >> 10, i = idx & 1023u;
  WT[idx] = (__bf16)W[(size_t)i * DM + o];
}

__global__ __launch_bounds__(256) void fill_mem_slots(const float* __restrict__ m_k,
                                                      const float* __restrict__ m_v,
                                                      __bf16* __restrict__ Kb,
                                                      __bf16* __restrict__ Vb) {
  unsigned int idx = blockIdx.x * 256u + threadIdx.x;  // B*M*DM = 131072
  if (idx >= (unsigned)(B_ * MSLOT * DM)) return;
  int b = idx >> 14;
  int rem = idx & 16383;
  int m = rem >> 10, j = rem & 1023;
  size_t row = (size_t)b * NKM + NK + m;
  Kb[row * DM + j] = (__bf16)(11.313708498984761f * m_k[(size_t)m * DM + j]);
  Vb[row * DM + j] = (__bf16)(4.0f * m_v[(size_t)m * DM + j]);
}

__global__ __launch_bounds__(256) void build_vt(const __bf16* __restrict__ Vb,
                                                __bf16* __restrict__ Vt) {
  size_t idx = (size_t)blockIdx.x * 256 + threadIdx.x;  // 64*128*1056
  if (idx >= (size_t)B_ * H_ * DK * VTLD) return;
  int bh = (int)(idx / (DK * VTLD));
  int rem = (int)(idx % (DK * VTLD));
  int d = rem / VTLD, k = rem % VTLD;
  int b = bh >> 3, h = bh & 7;
  __bf16 v = (__bf16)0.0f;
  if (k < NKM) v = Vb[((size_t)b * NKM + k) * DM + h * DK + d];
  Vt[idx] = v;
}

// ---------------------------------------------------------------------------
// WMMA GEMM: out(Mx1024) = A(Mx1024) @ BT^T + bias (BT = B^T, 1024x1024).
// 8 waves -> 128x64 block tile, 32x32 per wave, K steps of 32.
// ---------------------------------------------------------------------------
template <bool OUT_BF16>
__global__ __launch_bounds__(256) void gemm_bf16(const __bf16* __restrict__ A,
                                                 const __bf16* __restrict__ BT,
                                                 const float* __restrict__ bias,
                                                 void* __restrict__ out,
                                                 int rows_per_batch_out) {
  const int lane = threadIdx.x & 31;
  const int wave = threadIdx.x >> 5;
  const int row0 = blockIdx.x * 128 + (wave & 3) * 32;
  const int col0 = blockIdx.y * 64 + (wave >> 2) * 32;

  v8f c[2][2] = {};
  for (int k0 = 0; k0 < DM; k0 += 32) {
    v16bf a0 = frag_ld(A, DM, row0, 0x3FFFFFFF, k0);
    v16bf a1 = frag_ld(A, DM, row0 + 16, 0x3FFFFFFF, k0);
    v16bf b0 = frag_ld(BT, DM, col0, 0x3FFFFFFF, k0);
    v16bf b1 = frag_ld(BT, DM, col0 + 16, 0x3FFFFFFF, k0);
    c[0][0] = wmma_bf16(a0, b0, c[0][0]);
    c[0][1] = wmma_bf16(a0, b1, c[0][1]);
    c[1][0] = wmma_bf16(a1, b0, c[1][0]);
    c[1][1] = wmma_bf16(a1, b1, c[1][1]);
  }

  const int hi8 = (lane & 16) ? 8 : 0;
  const int cn = lane & 15;
#pragma unroll
  for (int mi = 0; mi < 2; ++mi)
#pragma unroll
    for (int ni = 0; ni < 2; ++ni)
#pragma unroll
      for (int r = 0; r < 8; ++r) {
        int m = row0 + mi * 16 + r + hi8;
        int n = col0 + ni * 16 + cn;
        float v = c[mi][ni][r] + bias[n];
        int b = m >> 10;
        int rr = m & 1023;
        size_t orow = (size_t)b * rows_per_batch_out + rr;
        if constexpr (OUT_BF16)
          ((__bf16*)out)[orow * DM + n] = (__bf16)v;
        else
          ((float*)out)[orow * DM + n] = v;
      }
}

// ---------------------------------------------------------------------------
// Fused attention: per (b,h, 32-query strip):
//   phase 1: stage Q tile (32x128 bf16) to LDS (async-to-LDS if available),
//            each wave computes 32x64 score chunks (17 round-robin k-tiles),
//            scale + mask -> fp32 logits in LDS strip (32 x 1040).
//   phase 2: per-wave row softmax (shuffle reductions), normalized probs
//            written once to global att (fp32) and kept in LDS.
//   phase 3: att@V: A frags from LDS probs (fp32->bf16), B frags from V^T,
//            each wave owns a 16-wide d slice; bf16 O tile out.
// Grid: (B*H, NQ/32). Block: 256 threads. Dynamic LDS: 144000 B.
// ---------------------------------------------------------------------------
__global__ __launch_bounds__(256) void attn_fused(const __bf16* __restrict__ Qb,
                                                  const __bf16* __restrict__ Kb,
                                                  const __bf16* __restrict__ Vt,
                                                  const unsigned char* __restrict__ mask,
                                                  float* __restrict__ att,
                                                  __bf16* __restrict__ Ob) {
  extern __shared__ __align__(16) char smem[];
  float*  sS = (float*)smem;                         // 32 x SSTRIDE fp32 logits/probs
  __bf16* sQ = (__bf16*)(smem + SMEM_S_BYTES);       // 32 x QSTRIDE bf16 Q tile

  const int bh = blockIdx.x;
  const int b = bh >> 3, h = bh & 7;
  const int q0 = blockIdx.y * QTILE;
  const int tid = threadIdx.x;
  const int lane = tid & 31;
  const int wave = tid >> 5;

  const __bf16* Qh = Qb + (size_t)b * NQ * DM + h * DK;
  const __bf16* Kh = Kb + (size_t)b * NKM * DM + h * DK;
  const __bf16* vt = Vt + (size_t)bh * DK * VTLD;

  // ---- stage Q tile (32 rows x 128 cols) into LDS: 512 x b128 transfers ----
  for (int v = tid; v < QTILE * (DK / 8); v += 256) {   // 512 vectors of 8 bf16
    int row = v >> 4, cv = v & 15;
    const __bf16* src = Qh + (size_t)(q0 + row) * DM + cv * 8;
    __bf16* dst = &sQ[row * QSTRIDE + cv * 8];
#if defined(HAVE_ASYNC_LDS)
    v4i* srcg = (v4i*)src;             // reinterpret (generic AS)
    v4i* dstg = (v4i*)dst;
    __builtin_amdgcn_global_load_async_to_lds_b128(
        (as1_v4i*)srcg, (as3_v4i*)dstg, 0, 0);
#else
    *reinterpret_cast<uint4*>(dst) = *reinterpret_cast<const uint4*>(src);
#endif
  }
#if defined(HAVE_ASYNC_LDS)
  __builtin_amdgcn_s_wait_asynccnt(0);
#endif
  __syncthreads();

  // ---- phase 1: scores into LDS strip ----
  const float scale = 0.088388347648318447f;  // 1/sqrt(128)
  const int hi8 = (lane & 16) ? 8 : 0;
  const int cn = lane & 15;
  const int ntiles = (NKM + 63) / 64;         // 17

  for (int t = wave; t < ntiles; t += 8) {
    const int k0 = t * 64;
    v8f c[2][4] = {};
#pragma unroll
    for (int d0 = 0; d0 < DK; d0 += 32) {
      v16bf a0 = frag_ld(sQ, QSTRIDE, 0, QTILE - 1, d0);
      v16bf a1 = frag_ld(sQ, QSTRIDE, 16, QTILE - 1, d0);
#pragma unroll
      for (int j = 0; j < 4; ++j) {
        v16bf bf = frag_ld(Kh, DM, k0 + j * 16, NKM - 1, d0);  // clamp rows>=1040
        c[0][j] = wmma_bf16(a0, bf, c[0][j]);
        c[1][j] = wmma_bf16(a1, bf, c[1][j]);
      }
    }
#pragma unroll
    for (int mi = 0; mi < 2; ++mi)
#pragma unroll
      for (int j = 0; j < 4; ++j)
#pragma unroll
        for (int r = 0; r < 8; ++r) {
          int qr = mi * 16 + r + hi8;        // row within tile
          int k = k0 + j * 16 + cn;
          if (k < NKM) {
            float v = c[mi][j][r] * scale;
            int gq = q0 + qr;
            if (k < NK && mask[((size_t)bh * NQ + gq) * NK + k])
              v = -__builtin_inff();
            sS[qr * SSTRIDE + k] = v;
          }
        }
  }
  __syncthreads();

  // ---- phase 2: softmax per row; write probs to global att once ----
  for (int r = 0; r < 4; ++r) {
    const int qr = wave * 4 + r;
    float* row = sS + qr * SSTRIDE;
    float mx = -__builtin_inff();
    for (int i = lane; i < NKM; i += 32) mx = fmaxf(mx, row[i]);
#pragma unroll
    for (int off = 16; off > 0; off >>= 1) mx = fmaxf(mx, __shfl_xor(mx, off, 32));
    float sum = 0.0f;
    for (int i = lane; i < NKM; i += 32) {
      float e = __expf(row[i] - mx);
      row[i] = e;
      sum += e;
    }
#pragma unroll
    for (int off = 16; off > 0; off >>= 1) sum += __shfl_xor(sum, off, 32);
    const float inv = 1.0f / sum;
    float* grow = att + ((size_t)bh * NQ + (q0 + qr)) * NKM;
    for (int i = lane; i < NKM; i += 32) {
      float p = row[i] * inv;
      row[i] = p;        // keep in LDS for phase 3
      grow[i] = p;       // single global write of att
    }
  }
  __syncthreads();

  // ---- phase 3: O tile = probs @ V_h; wave owns 16-wide d slice ----
  const int d0 = wave * 16;
  v8f o[2] = {};
  for (int k0 = 0; k0 < NKM; k0 += 32) {
    const bool tail = (k0 + 32 > NKM);
    v16bf a0 = frag_ld_f32(sS, SSTRIDE, 0, k0, tail);
    v16bf a1 = frag_ld_f32(sS, SSTRIDE, 16, k0, tail);
    v16bf bf = frag_ld(vt, VTLD, d0, DK - 1, k0);   // padded zeros past 1040
    o[0] = wmma_bf16(a0, bf, o[0]);
    o[1] = wmma_bf16(a1, bf, o[1]);
  }
#pragma unroll
  for (int mi = 0; mi < 2; ++mi)
#pragma unroll
    for (int r = 0; r < 8; ++r) {
      int q = q0 + mi * 16 + r + hi8;
      Ob[((size_t)b * NQ + q) * DM + h * DK + d0 + cn] = (__bf16)o[mi][r];
    }
}

// ---------------------------------------------------------------------------
extern "C" void kernel_launch(void* const* d_in, const int* in_sizes, int n_in,
                              void* d_out, int out_size, void* d_ws, size_t ws_size,
                              hipStream_t stream) {
  (void)in_sizes; (void)n_in; (void)out_size; (void)ws_size;

  const float* queries = (const float*)d_in[0];
  const float* keys    = (const float*)d_in[1];
  const float* values  = (const float*)d_in[2];
  const unsigned char* mask = (const unsigned char*)d_in[3];
  const float* Wq = (const float*)d_in[4];
  const float* bq = (const float*)d_in[5];
  const float* Wk = (const float*)d_in[6];
  const float* bk = (const float*)d_in[7];
  const float* Wv = (const float*)d_in[8];
  const float* bv = (const float*)d_in[9];
  const float* Wo = (const float*)d_in[10];
  const float* bo = (const float*)d_in[11];
  const float* m_k = (const float*)d_in[12];
  const float* m_v = (const float*)d_in[13];

  char* ws = (char*)d_ws;
  size_t off = 0;
  auto alloc_bf16 = [&](size_t elems) {
    __bf16* p = (__bf16*)(ws + off);
    off = (off + elems * sizeof(__bf16) + 255) & ~(size_t)255;
    return p;
  };
  const size_t NACT = (size_t)B_ * NQ * DM;        // 8388608
  const size_t NW   = (size_t)DM * DM;             // 1048576
  const size_t NKV  = (size_t)B_ * NKM * DM;       // 8519680
  const size_t NVT  = (size_t)B_ * H_ * DK * VTLD; // 8650752

  __bf16* Xq  = alloc_bf16(NACT);
  __bf16* Xk  = alloc_bf16(NACT);
  __bf16* Xv  = alloc_bf16(NACT);
  __bf16* WqT = alloc_bf16(NW);
  __bf16* WkT = alloc_bf16(NW);
  __bf16* WvT = alloc_bf16(NW);
  __bf16* WoT = alloc_bf16(NW);
  __bf16* Qb  = alloc_bf16(NACT);
  __bf16* Kb  = alloc_bf16(NKV);
  __bf16* Vb  = alloc_bf16(NKV);
  __bf16* Vt  = alloc_bf16(NVT);
  __bf16* Ob  = alloc_bf16(NACT);

  float* out_f = (float*)d_out;                    // (B, NQ, DM)
  float* att   = out_f + NACT;                     // (B*H, NQ, NKM) fp32

  cvt_f32_bf16<<<32768, 256, 0, stream>>>(queries, Xq, (unsigned)NACT);
  cvt_f32_bf16<<<32768, 256, 0, stream>>>(keys, Xk, (unsigned)NACT);
  cvt_f32_bf16<<<32768, 256, 0, stream>>>(values, Xv, (unsigned)NACT);

  wt_cvt<<<4096, 256, 0, stream>>>(Wq, WqT);
  wt_cvt<<<4096, 256, 0, stream>>>(Wk, WkT);
  wt_cvt<<<4096, 256, 0, stream>>>(Wv, WvT);
  wt_cvt<<<4096, 256, 0, stream>>>(Wo, WoT);

  dim3 gproj(64, 16, 1);
  gemm_bf16<true><<<gproj, 256, 0, stream>>>(Xq, WqT, bq, Qb, NQ);
  gemm_bf16<true><<<gproj, 256, 0, stream>>>(Xk, WkT, bk, Kb, NKM);
  gemm_bf16<true><<<gproj, 256, 0, stream>>>(Xv, WvT, bv, Vb, NKM);

  fill_mem_slots<<<512, 256, 0, stream>>>(m_k, m_v, Kb, Vb);
  build_vt<<<(unsigned)((NVT + 255) / 256), 256, 0, stream>>>(Vb, Vt);

  // fused scores + softmax + att@V : att written exactly once
  attn_fused<<<dim3(B_ * H_, NQ / QTILE), 256, SMEM_BYTES, stream>>>(
      Qb, Kb, Vt, mask, att, Ob);

  gemm_bf16<false><<<gproj, 256, 0, stream>>>(Ob, WoT, bo, out_f, NQ);
}
